// ColorAttention_25486335935155
// MI455X (gfx1250) — matrix operand order repaired and checked by reading, not compiled, not run on hardware
//
#include <hip/hip_runtime.h>
#include <math.h>

#define E_DIM 512
#define NHEAD 8
#define HDIM  64
#define SEQL  256
#define NCLS  313
#define N1T   569
#define NPAD  576
#define BATCH 8
#define IMG   256
#define PATCH 16

typedef __bf16 bf16_t;
typedef __attribute__((ext_vector_type(16))) __bf16 v16bf;
typedef __attribute__((ext_vector_type(8)))  float  v8f;

// workspace layout (bytes)
#define ADDM_OFF 0u            // 8*256*313*4  = 2,564,096
#define Q_OFF    (4u << 20)    // 8*8*576*64*2 = 4,718,592 each
#define K_OFF    (9u << 20)
#define V_OFF    (14u << 20)   // vT: [b][h][d][576] bf16, same total size
#define AO_OFF   (19u << 20)   // 8*576*512*4  = 9,437,184

// ---------------------------------------------------------------------------
// Kernel 1: mask patch-reduction (bandwidth bound: reads 656 MB).
// One block per (b, patch); lane = color class (innermost contiguous dim).
// addm[b][s][c] = (sum over 16x16 pixels == 1.0f) ? 0 : -inf
// ---------------------------------------------------------------------------
__global__ void k_mask_reduce(const float* __restrict__ mask,
                              float* __restrict__ addm) {
    int blk = blockIdx.x;               // b*SEQL + s
    int b = blk / SEQL, s = blk % SEQL;
    int ph = s / (IMG / PATCH), pw = s % (IMG / PATCH);
    int c = threadIdx.x;
    if (c >= NCLS) return;
    const float* base = mask + (((size_t)b * IMG + ph * PATCH) * IMG + pw * PATCH) * NCLS + c;
    float sum = 0.0f;
    for (int p1 = 0; p1 < PATCH; ++p1) {
        const float* rowp = base + (size_t)p1 * IMG * NCLS;
        __builtin_prefetch(rowp + (size_t)IMG * NCLS, 0, 1);
        #pragma unroll 4
        for (int p2 = 0; p2 < PATCH; ++p2) sum += rowp[(size_t)p2 * NCLS];
    }
    addm[(size_t)(b * SEQL + s) * NCLS + c] = (sum == 1.0f) ? 0.0f : -__builtin_inff();
}

// ---------------------------------------------------------------------------
// Kernel 2: QKV projection with v_wmma_f32_16x16x32_bf16.
// One wave computes one 16x16 tile of [576 x 1536]; K = 512 in 16 steps.
// q is scaled by 1/tau at store; rows >= 569 stored as zero (padding).
// q,k stored [b][h][n][d]; v stored TRANSPOSED [b][h][d][n] for phase-3 B loads.
// ---------------------------------------------------------------------------
__global__ void k_qkv(const float* __restrict__ x, const float* __restrict__ colors,
                      const float* __restrict__ qkv_w, const float* __restrict__ tau_p,
                      bf16_t* __restrict__ qb, bf16_t* __restrict__ kb,
                      bf16_t* __restrict__ vt) {
    int mt = blockIdx.x, nt = blockIdx.y, b = blockIdx.z;
    int lane = threadIdx.x;
    int r16 = lane & 15, lh = lane >> 4;
    int row = mt * 16 + r16;            // A row (token)
    int col = nt * 16 + r16;            // B col (output feature in [0,1536))

    const float* arow = nullptr;
    if (row < SEQL)      arow = x + (size_t)(b * SEQL + row) * E_DIM;
    else if (row < N1T)  arow = colors + (size_t)(b * NCLS + (row - SEQL)) * E_DIM;
    const float* brow = qkv_w + (size_t)col * E_DIM;

    v8f acc = {0.f,0.f,0.f,0.f,0.f,0.f,0.f,0.f};
    int h8 = lh * 8, h16 = lh * 16;
    for (int kc = 0; kc < E_DIM; kc += 32) {
        v16bf a, bb;
        if (arow) {
            const float* p0 = arow + kc + h16;
            const float* p1 = arow + kc + h16 + 8;
            #pragma unroll
            for (int e = 0; e < 8; ++e) { a[e] = (bf16_t)p0[e]; a[8 + e] = (bf16_t)p1[e]; }
        } else {
            #pragma unroll
            for (int e = 0; e < 16; ++e) a[e] = (bf16_t)0.0f;
        }
        const float* pb = brow + kc + h16;
        #pragma unroll
        for (int e = 0; e < 16; ++e) bb[e] = (bf16_t)pb[e];
        acc = __builtin_amdgcn_wmma_f32_16x16x32_bf16(false, a, false, bb,
                                                      (short)0, acc, false, false);
    }

    int which = col >> 9;               // 0=q, 1=k, 2=v
    int hf = col & 511;
    int head = hf >> 6, d = hf & 63;
    float scale = (which == 0) ? (1.0f / tau_p[0]) : 1.0f;
    size_t hb = (size_t)(b * NHEAD + head);
    #pragma unroll
    for (int j = 0; j < 8; ++j) {
        int n = mt * 16 + j + lh * 8;   // D row layout
        float val = (n < N1T) ? acc[j] * scale : 0.0f;
        if (which == 2) {
            // transposed: vT[b][h][d][n]  (consecutive j -> contiguous n)
            vt[(hb * HDIM + d) * NPAD + n] = (bf16_t)val;
        } else {
            bf16_t* dst = (which == 0) ? qb : kb;
            dst[(hb * NPAD + n) * HDIM + d] = (bf16_t)val;
        }
    }
}

// ---------------------------------------------------------------------------
// Kernel 3: fused attention per (b, head, 16-row q-tile).
// scores (16x576, WMMA qk^T) -> +mask -> softmax in LDS -> out = attn @ vT.
// ---------------------------------------------------------------------------
__global__ void k_attn(const bf16_t* __restrict__ qb, const bf16_t* __restrict__ kb,
                       const bf16_t* __restrict__ vt, const float* __restrict__ addm,
                       float* __restrict__ ao) {
    __shared__ float sc[16][NPAD + 4];   // +4 keeps rows 16B-aligned, shifts banks
    int qt = blockIdx.x, h = blockIdx.y, b = blockIdx.z;
    int lane = threadIdx.x;
    int r16 = lane & 15, lh = lane >> 4;
    int h8 = lh * 8, h16 = lh * 16;
    size_t headoff = (size_t)(b * NHEAD + h) * NPAD;   // rows base for q/k
    size_t vtoff   = (size_t)(b * NHEAD + h) * HDIM;   // d base for vT

    // A fragments of q (d-chunks 0..31 and 32..63)
    const bf16_t* qp = qb + (headoff + qt * 16 + r16) * HDIM;
    v16bf a0, a1;
    #pragma unroll
    for (int e = 0; e < 8; ++e) {
        a0[e] = qp[h16 + e];        a0[8 + e] = qp[h16 + 8 + e];
        a1[e] = qp[32 + h16 + e];   a1[8 + e] = qp[40 + h16 + e];
    }

    // Phase 1: scores row-block into LDS
    for (int mtile = 0; mtile < NPAD / 16; ++mtile) {
        const bf16_t* kp = kb + (headoff + mtile * 16 + r16) * HDIM;
        v16bf b0, b1;
        #pragma unroll
        for (int e = 0; e < 16; ++e) { b0[e] = kp[h16 + e]; b1[e] = kp[32 + h16 + e]; }
        v8f dacc = {0.f,0.f,0.f,0.f,0.f,0.f,0.f,0.f};
        dacc = __builtin_amdgcn_wmma_f32_16x16x32_bf16(false, a0, false, b0,
                                                       (short)0, dacc, false, false);
        dacc = __builtin_amdgcn_wmma_f32_16x16x32_bf16(false, a1, false, b1,
                                                       (short)0, dacc, false, false);
        #pragma unroll
        for (int j = 0; j < 8; ++j) {
            int r = j + lh * 8;
            int n = qt * 16 + r;
            int m = mtile * 16 + r16;
            float mv = 0.0f;
            if (m >= N1T) mv = -__builtin_inff();
            else if (n < N1T) {
                if (n < SEQL && m >= SEQL)
                    mv = addm[((size_t)b * SEQL + n) * NCLS + (m - SEQL)];
                else if (n >= SEQL && m < SEQL)
                    mv = addm[((size_t)b * SEQL + m) * NCLS + (n - SEQL)];
            }
            sc[r][m] = dacc[j] + mv;
        }
    }
    __syncthreads();

    // Phase 2: softmax; two lanes share a row, each covers 288 columns
    {
        int row = r16;
        int c0 = lh * (NPAD / 2);
        float mx = -__builtin_inff();
        for (int c = 0; c < NPAD / 2; ++c) mx = fmaxf(mx, sc[row][c0 + c]);
        mx = fmaxf(mx, __shfl_xor(mx, 16));
        float sum = 0.0f;
        for (int c = 0; c < NPAD / 2; ++c) {
            float ev = __expf(sc[row][c0 + c] - mx);
            sc[row][c0 + c] = ev;
            sum += ev;
        }
        sum += __shfl_xor(sum, 16);
        float inv = 1.0f / sum;
        for (int c = 0; c < NPAD / 2; ++c) sc[row][c0 + c] *= inv;
    }
    __syncthreads();

    // Phase 3: out tile = attn(16x576) @ v(576x64), four 16-col tiles.
    // B fragment now reads vT rows: 16 contiguous bf16 per lane per WMMA.
    #pragma unroll
    for (int nt2 = 0; nt2 < 4; ++nt2) {
        int d = nt2 * 16 + r16;                      // B col (head dim)
        const bf16_t* vp = vt + (vtoff + d) * NPAD;  // contiguous over m
        v8f acc = {0.f,0.f,0.f,0.f,0.f,0.f,0.f,0.f};
        for (int kc = 0; kc < NPAD; kc += 32) {
            v16bf a, bv;
            #pragma unroll
            for (int e = 0; e < 8; ++e) {
                a[e]     = (bf16_t)sc[r16][kc + h16 + e];
                a[8 + e] = (bf16_t)sc[r16][kc + h16 + 8 + e];
            }
            const bf16_t* pv = vp + kc + h16;
            #pragma unroll
            for (int e = 0; e < 16; ++e) bv[e] = pv[e];
            acc = __builtin_amdgcn_wmma_f32_16x16x32_bf16(false, a, false, bv,
                                                          (short)0, acc, false, false);
        }
        #pragma unroll
        for (int j = 0; j < 8; ++j) {
            int n = qt * 16 + j + lh * 8;
            int e = h * HDIM + nt2 * 16 + r16;
            ao[((size_t)b * NPAD + n) * E_DIM + e] = acc[j];
        }
    }
}

// ---------------------------------------------------------------------------
// Kernel 4: output projection [576 x 512] @ o_w^T + o_b -> d_out (569 rows)
// ---------------------------------------------------------------------------
__global__ void k_oproj(const float* __restrict__ ao, const float* __restrict__ o_w,
                        const float* __restrict__ o_b, float* __restrict__ out) {
    int mt = blockIdx.x, nt = blockIdx.y, b = blockIdx.z;
    int lane = threadIdx.x;
    int r16 = lane & 15, lh = lane >> 4;
    int h8 = lh * 8, h16 = lh * 16;
    const float* arow = ao + ((size_t)b * NPAD + mt * 16 + r16) * E_DIM;
    const float* brow = o_w + (size_t)(nt * 16 + r16) * E_DIM;
    v8f acc = {0.f,0.f,0.f,0.f,0.f,0.f,0.f,0.f};
    for (int kc = 0; kc < E_DIM; kc += 32) {
        v16bf a, bb;
        const float* p0 = arow + kc + h16;
        const float* p1 = arow + kc + h16 + 8;
        #pragma unroll
        for (int e = 0; e < 8; ++e) { a[e] = (bf16_t)p0[e]; a[8 + e] = (bf16_t)p1[e]; }
        const float* pb = brow + kc + h16;
        #pragma unroll
        for (int e = 0; e < 16; ++e) bb[e] = (bf16_t)pb[e];
        acc = __builtin_amdgcn_wmma_f32_16x16x32_bf16(false, a, false, bb,
                                                      (short)0, acc, false, false);
    }
    int f = nt * 16 + r16;
    float bias = o_b[f];
    #pragma unroll
    for (int j = 0; j < 8; ++j) {
        int n = mt * 16 + j + lh * 8;
        if (n < N1T) out[((size_t)(b * N1T) + n) * E_DIM + f] = acc[j] + bias;
    }
}

extern "C" void kernel_launch(void* const* d_in, const int* in_sizes, int n_in,
                              void* d_out, int out_size, void* d_ws, size_t ws_size,
                              hipStream_t stream) {
    const float* x      = (const float*)d_in[0];
    const float* colors = (const float*)d_in[1];
    const float* mask   = (const float*)d_in[2];
    const float* qkv_w  = (const float*)d_in[3];
    const float* o_w    = (const float*)d_in[4];
    const float* o_b    = (const float*)d_in[5];
    const float* tau    = (const float*)d_in[6];

    char* ws = (char*)d_ws;
    float*   addm = (float*)(ws + ADDM_OFF);
    bf16_t*  qb   = (bf16_t*)(ws + Q_OFF);
    bf16_t*  kb   = (bf16_t*)(ws + K_OFF);
    bf16_t*  vt   = (bf16_t*)(ws + V_OFF);
    float*   ao   = (float*)(ws + AO_OFF);
    float*   out  = (float*)d_out;

    // 1) mask reduction: 2048 blocks x 320 lanes (10 waves), coalesced over classes
    k_mask_reduce<<<dim3(BATCH * SEQL), dim3(320), 0, stream>>>(mask, addm);
    // 2) QKV GEMM: [576 x 1536] per batch, one wave per 16x16 tile
    k_qkv<<<dim3(NPAD / 16, 96, BATCH), dim3(32), 0, stream>>>(x, colors, qkv_w, tau,
                                                               qb, kb, vt);
    // 3) fused attention per (b, head, q-tile)
    k_attn<<<dim3(NPAD / 16, NHEAD, BATCH), dim3(32), 0, stream>>>(qb, kb, vt, addm, ao);
    // 4) output projection + bias
    k_oproj<<<dim3(NPAD / 16, E_DIM / 16, BATCH), dim3(32), 0, stream>>>(ao, o_w, o_b, out);
}
